// DynamicsHydrogen_71640054497690
// MI455X (gfx1250) — compile-verified, matrix operand order checked by reading, not run on hardware
//
#include <hip/hip_runtime.h>
#include <hip/hip_bf16.h>

typedef __attribute__((ext_vector_type(16))) _Float16 v16h;
typedef __attribute__((ext_vector_type(8)))  _Float16 v8h;
typedef __attribute__((ext_vector_type(8)))  float    v8f;

#define B_    8
#define S_    4
#define H_    50
#define NH_   200
#define EHH_  400
#define L_    2
#define HID_  128
#define FDIM_ 79
#define EDIM_ 161
#define NNODE_ 250
#define FSTR_ 96                 // padded feature row (f16 elems)
#define E_    (EHH_ + H_*NH_)    // 10400
#define KP1_  192                // padded K for edge MLP input
#define KPN_  224                // padded K for node MLP input
#define EFS_  (16*KP1_)          // per-wave LDS halves (edge kernel)
#define NFS_  (16*KPN_)          // per-wave LDS halves (node kernel)

__device__ __forceinline__ float siluf(float x) {
    return x * (1.0f / (1.0f + __expf(-x)));
}

// A-fragment (16x32 f16, M x K) from an LDS row-major buffer with row stride ldk halves.
// Per ISA: lanes 0-15 hold K {k0..k0+7, k0+16..k0+23}; lanes 16-31 hold {k0+8..15, k0+24..31}.
__device__ __forceinline__ v16h load_a_frag(const _Float16* buf, int ldk, int k0) {
    int lane = threadIdx.x & 31;
    int r = lane & 15, hi = lane >> 4;
    const _Float16* p = buf + r * ldk + k0 + hi * 8;
    v8h lo = *(const v8h*)(p);
    v8h hh = *(const v8h*)(p + 16);
    v16h a;
#pragma unroll
    for (int i = 0; i < 8; ++i) { a[i] = lo[i]; a[8 + i] = hh[i]; }
    return a;
}

// B-fragment from pre-swizzled global memory (frag = 32 lanes x 16 halves contiguous).
__device__ __forceinline__ v16h load_b_frag(const _Float16* frag) {
    const _Float16* p = frag + (threadIdx.x & 31) * 16;
    v8h lo = *(const v8h*)(p);
    v8h hh = *(const v8h*)(p + 8);
    v16h b;
#pragma unroll
    for (int i = 0; i < 8; ++i) { b[i] = lo[i]; b[8 + i] = hh[i]; }
    return b;
}

__device__ __forceinline__ v8f wmma_f16(v16h a, v16h b, v8f c) {
    return __builtin_amdgcn_wmma_f32_16x16x32_f16(false, a, false, b, (short)0, c,
                                                  false, false);
}

__device__ __forceinline__ void atomAddF(float* p, float v) {
    unsafeAtomicAdd(p, v);   // hardware global_atomic_add_f32
}

// ---------------------------------------------------------------------------
// Weight swizzle: f32 [K x N] row-major -> f16 WMMA B-fragments.
// mode 0: identity K (k < Kreal), mode 1: edge-feature K permutation (EDIM=161),
// mode 2: node-feature K permutation (FDIM+HID=207).
__global__ void swizzle_weights(const float* __restrict__ src, _Float16* __restrict__ dst,
                                int NT, int Nreal, int srcld, int mode, int Kreal) {
    int frag = blockIdx.x;
    int lane = threadIdx.x;
    int kt = frag / NT, nt = frag % NT;
    int nlo = lane & 15, hi = lane >> 4;
    int n = nt * 16 + nlo;
    _Float16* out = dst + (size_t)frag * 512 + lane * 16;
#pragma unroll
    for (int j = 0; j < 16; ++j) {
        int kl = hi * 8 + (j < 8 ? j : j + 8);
        int k = kt * 32 + kl;
        int km;
        if (mode == 0) {
            km = (k < Kreal) ? k : -1;
        } else if (mode == 1) {     // ef layout: [0..78]=hi, [96..174]=hj, 176=dist, 177=bond, 178=t
            if (k < 79) km = k;
            else if (k < 96) km = -1;
            else if (k < 175) km = k - 17;      // -> 79..157
            else if (k == 176) km = 158;
            else if (k == 177) km = 159;
            else if (k == 178) km = 160;
            else km = -1;
        } else {                    // nf layout: [0..78]=h_h, [96..223]=agg_m
            if (k < 79) km = k;
            else if (k < 96) km = -1;
            else km = k - 17;                   // -> 79..206
        }
        float v = (km >= 0 && n < Nreal) ? src[(size_t)km * srcld + n] : 0.0f;
        out[j] = (_Float16)v;
    }
}

// ---------------------------------------------------------------------------
__global__ void prep_feat(const int* __restrict__ lab_h, const int* __restrict__ lab_hv,
                          const int* __restrict__ pos_h, const int* __restrict__ pos_hv,
                          const int* __restrict__ pep, _Float16* __restrict__ feat) {
    int idx = blockIdx.x * blockDim.x + threadIdx.x;
    if (idx >= B_ * NNODE_) return;
    int b = idx / NNODE_, node = idx % NNODE_;
    int label, pos;
    if (node < H_) { label = lab_h[b * H_ + node];        pos = pos_h[b * H_ + node]; }
    else           { label = lab_hv[b * NH_ + node - H_]; pos = pos_hv[b * NH_ + node - H_]; }
    int aa = pep[b * 15 + (pos - 1)];
    _Float16 row[FSTR_];
#pragma unroll
    for (int i = 0; i < FSTR_; ++i) row[i] = (_Float16)0.0f;
    row[label] = (_Float16)1.0f;
    row[44 + aa] = (_Float16)1.0f;
    row[64 + pos - 1] = (_Float16)1.0f;
    for (int s = 0; s < S_; ++s) {
        _Float16* d = feat + ((size_t)(b * S_ + s) * NNODE_ + node) * FSTR_;
        for (int i = 0; i < FSTR_; ++i) d[i] = row[i];
    }
}

__global__ void prep_edges(const int* __restrict__ ehh, const float* __restrict__ emhh,
                           const float* __restrict__ emhv, const float* __restrict__ bondm,
                           int* __restrict__ erow, int* __restrict__ ecol,
                           float* __restrict__ ebond, float* __restrict__ emask) {
    int idx = blockIdx.x * blockDim.x + threadIdx.x;
    if (idx >= B_ * E_) return;
    int b = idx / E_, e = idx % E_;
    int r, c; float bd, em;
    if (e < EHH_) {
        r = ehh[(b * EHH_ + e) * 2 + 0];
        c = ehh[(b * EHH_ + e) * 2 + 1];
        em = emhh[b * EHH_ + e];
        bd = 0.0f;                         // H-H block of bond matrix is zero
    } else {
        int g = e - EHH_;
        r = g / NH_;
        int cc = g % NH_;
        c = H_ + cc;
        em = emhv[(b * H_ + r) * NH_ + cc];
        bd = bondm[(b * H_ + r) * NH_ + cc];
    }
    erow[idx] = r; ecol[idx] = c; ebond[idx] = bd; emask[idx] = em;
}

__global__ void prep_x(const float* __restrict__ xh, const float* __restrict__ xhv,
                       const int* __restrict__ bound, float* __restrict__ x,
                       float* __restrict__ x0) {
    int idx = blockIdx.x * blockDim.x + threadIdx.x;
    if (idx >= B_ * S_ * NNODE_) return;
    int node = idx % NNODE_;
    int bs = idx / NNODE_;
    int b = bs / S_;
    if (node < H_) {
        int ba = bound[b * H_ + node];
#pragma unroll
        for (int d = 0; d < 3; ++d) {
            float p = xh[((size_t)bs * H_ + node) * 3 + d] + xhv[((size_t)b * NH_ + ba) * 3 + d];
            x[((size_t)bs * NNODE_ + node) * 3 + d] = p;
            x0[((size_t)bs * H_ + node) * 3 + d] = p;
        }
    } else {
#pragma unroll
        for (int d = 0; d < 3; ++d)
            x[((size_t)bs * NNODE_ + node) * 3 + d] = xhv[((size_t)b * NH_ + node - H_) * 3 + d];
    }
}

// ---------------------------------------------------------------------------
// Edge MLP: one wave = 16 edges. GEMM1(192->128) -> GEMM2(128->128) -> GEMM3(128->128)
// -> dot(128->1); scatter agg_m / agg_x with f32 atomics.
__global__ void __launch_bounds__(256)
edge_kernel(const _Float16* __restrict__ feat, const float* __restrict__ x,
            const int* __restrict__ erow, const int* __restrict__ ecol,
            const float* __restrict__ ebond, const float* __restrict__ emaskg,
            const float* __restrict__ tptr,
            const _Float16* __restrict__ w1f, const _Float16* __restrict__ w2f,
            const _Float16* __restrict__ wc1f,
            const float* __restrict__ b1, const float* __restrict__ b2,
            const float* __restrict__ bc1, const float* __restrict__ wc2,
            float* __restrict__ aggm, float* __restrict__ aggx) {
    __shared__ _Float16 smem[8 * EFS_];
    int bs = blockIdx.y;
    int b = bs >> 2;
    int w = threadIdx.x >> 5;
    int tile = blockIdx.x * 8 + w;
    if (tile >= E_ / 16) return;
    int e0 = tile * 16;
    int lane = threadIdx.x & 31;
    _Float16* ef = smem + w * EFS_;

    const _Float16* feat_bs = feat + (size_t)bs * NNODE_ * FSTR_;
    const float* x_bs = x + (size_t)bs * NNODE_ * 3;
    const int* erow_b = erow + (size_t)b * E_;
    const int* ecol_b = ecol + (size_t)b * E_;
    float* aggm_bs = aggm + (size_t)bs * H_ * HID_;
    float* aggx_bs = aggx + (size_t)bs * H_ * 3;

    // ---- build ef[16][192]: [0..78]=feat[row], [96..174]=feat[col], 176=dist,177=bond,178=t
    {
        int e = lane & 15;
        int part = lane >> 4;
        int row = erow_b[e0 + e];
        int col = ecol_b[e0 + e];
        int node = part ? col : row;
        const v8h* src = (const v8h*)(feat_bs + (size_t)node * FSTR_);
        v8h* dst = (v8h*)(ef + e * KP1_ + part * 96);
#pragma unroll
        for (int i = 0; i < 12; ++i) dst[i] = src[i];
        if (part) {
            float dx = x_bs[row * 3 + 0] - x_bs[col * 3 + 0];
            float dy = x_bs[row * 3 + 1] - x_bs[col * 3 + 1];
            float dz = x_bs[row * 3 + 2] - x_bs[col * 3 + 2];
            ef[e * KP1_ + 176] = (_Float16)sqrtf(dx * dx + dy * dy + dz * dz);
            ef[e * KP1_ + 177] = (_Float16)ebond[(size_t)b * E_ + e0 + e];
            ef[e * KP1_ + 178] = (_Float16)tptr[b];
        }
    }

    int hi = lane >> 4, nlo = lane & 15;
    int rowm[8], colm[8];
    float em[8];
#pragma unroll
    for (int r = 0; r < 8; ++r) {
        int e = e0 + r + 8 * hi;
        rowm[r] = erow_b[e];
        colm[r] = ecol_b[e];
        em[r] = emaskg[(size_t)b * E_ + e];
    }

    // ---- GEMM1: ef(16x192) @ W1(192x128)
    v8f acc[8];
#pragma unroll
    for (int nt = 0; nt < 8; ++nt)
#pragma unroll
        for (int i = 0; i < 8; ++i) acc[nt][i] = 0.0f;
#pragma unroll
    for (int kt = 0; kt < 6; ++kt) {
        v16h a = load_a_frag(ef, KP1_, kt * 32);
#pragma unroll
        for (int nt = 0; nt < 8; ++nt) {
            v16h bb = load_b_frag(w1f + (size_t)(kt * 8 + nt) * 512);
            acc[nt] = wmma_f16(a, bb, acc[nt]);
        }
    }
    // bias + SiLU -> stage m1 as f16 A-operand (reuse ef buffer, stride KP1_)
#pragma unroll
    for (int nt = 0; nt < 8; ++nt) {
        float bv = b1[nt * 16 + nlo];
#pragma unroll
        for (int r = 0; r < 8; ++r) {
            float v = siluf(acc[nt][r] + bv);
            ef[(r + 8 * hi) * KP1_ + nt * 16 + nlo] = (_Float16)v;
        }
    }

    // ---- GEMM2: m1 @ W2
    v8f acc2[8];
#pragma unroll
    for (int nt = 0; nt < 8; ++nt)
#pragma unroll
        for (int i = 0; i < 8; ++i) acc2[nt][i] = 0.0f;
#pragma unroll
    for (int kt = 0; kt < 4; ++kt) {
        v16h a = load_a_frag(ef, KP1_, kt * 32);
#pragma unroll
        for (int nt = 0; nt < 8; ++nt) {
            v16h bb = load_b_frag(w2f + (size_t)(kt * 8 + nt) * 512);
            acc2[nt] = wmma_f16(a, bb, acc2[nt]);
        }
    }
    // m2 = SiLU(.)*emask -> atomically scatter agg_m, stage m2 for GEMM3
#pragma unroll
    for (int nt = 0; nt < 8; ++nt) {
        float bv = b2[nt * 16 + nlo];
#pragma unroll
        for (int r = 0; r < 8; ++r) {
            float v = siluf(acc2[nt][r] + bv) * em[r];
            atomAddF(&aggm_bs[(size_t)rowm[r] * HID_ + nt * 16 + nlo], v);
            ef[(r + 8 * hi) * KP1_ + nt * 16 + nlo] = (_Float16)v;
        }
    }

    // ---- GEMM3: m2 @ Wc1, then cw = SiLU(.) @ Wc2
    v8f acc3[8];
#pragma unroll
    for (int nt = 0; nt < 8; ++nt)
#pragma unroll
        for (int i = 0; i < 8; ++i) acc3[nt][i] = 0.0f;
#pragma unroll
    for (int kt = 0; kt < 4; ++kt) {
        v16h a = load_a_frag(ef, KP1_, kt * 32);
#pragma unroll
        for (int nt = 0; nt < 8; ++nt) {
            v16h bb = load_b_frag(wc1f + (size_t)(kt * 8 + nt) * 512);
            acc3[nt] = wmma_f16(a, bb, acc3[nt]);
        }
    }
    float cwp[8];
#pragma unroll
    for (int r = 0; r < 8; ++r) cwp[r] = 0.0f;
#pragma unroll
    for (int nt = 0; nt < 8; ++nt) {
        float bv = bc1[nt * 16 + nlo];
        float wv = wc2[nt * 16 + nlo];
#pragma unroll
        for (int r = 0; r < 8; ++r) cwp[r] += siluf(acc3[nt][r] + bv) * wv;
    }
#pragma unroll
    for (int m = 1; m < 16; m <<= 1)
#pragma unroll
        for (int r = 0; r < 8; ++r) cwp[r] += __shfl_xor(cwp[r], m, 32);

    if (nlo < 3) {
        int d = nlo;
#pragma unroll
        for (int r = 0; r < 8; ++r) {
            float diff = x_bs[rowm[r] * 3 + d] - x_bs[colm[r] * 3 + d];
            atomAddF(&aggx_bs[rowm[r] * 3 + d], diff * cwp[r]);
        }
    }
}

// ---------------------------------------------------------------------------
// Node MLP: one wave = 16 nodes. nf(16x224) @ Wn1 -> SiLU -> @ Wn2; residual update.
__global__ void __launch_bounds__(128)
node_kernel(_Float16* __restrict__ feat, float* __restrict__ x,
            const float* __restrict__ aggm, const float* __restrict__ aggx,
            const _Float16* __restrict__ wn1f, const _Float16* __restrict__ wn2f,
            const float* __restrict__ bn1, const float* __restrict__ bn2) {
    __shared__ _Float16 smem[4 * NFS_];
    int bs = blockIdx.x;
    int w = threadIdx.x >> 5;
    int lane = threadIdx.x & 31;
    int base = w * 16;
    _Float16* nf = smem + w * NFS_;

    _Float16* feat_bs = feat + (size_t)bs * NNODE_ * FSTR_;
    const float* aggm_bs = aggm + (size_t)bs * H_ * HID_;
    const float* aggx_bs = aggx + (size_t)bs * H_ * 3;
    float* x_bs = x + (size_t)bs * NNODE_ * 3;

    // ---- build nf[16][224]: [0..78]=h_h (padded to 96), [96..223]=agg_m
    {
        int m = lane & 15, part = lane >> 4;
        int node = base + m;
        if (!part) {
            v8h* dst = (v8h*)(nf + m * KPN_);
            if (node < H_) {
                const v8h* src = (const v8h*)(feat_bs + (size_t)node * FSTR_);
#pragma unroll
                for (int i = 0; i < 12; ++i) dst[i] = src[i];
            } else {
                v8h z;
#pragma unroll
                for (int j = 0; j < 8; ++j) z[j] = (_Float16)0.0f;
#pragma unroll
                for (int i = 0; i < 12; ++i) dst[i] = z;
            }
        } else {
            _Float16* dst = nf + m * KPN_ + 96;
            if (node < H_) {
                const float* src = aggm_bs + (size_t)node * HID_;
                for (int i = 0; i < HID_; ++i) dst[i] = (_Float16)src[i];
            } else {
                for (int i = 0; i < HID_; ++i) dst[i] = (_Float16)0.0f;
            }
        }
    }

    int hi = lane >> 4, nlo = lane & 15;

    // ---- GEMM: nf @ Wn1 (224 -> 128)
    v8f acc[8];
#pragma unroll
    for (int nt = 0; nt < 8; ++nt)
#pragma unroll
        for (int i = 0; i < 8; ++i) acc[nt][i] = 0.0f;
#pragma unroll
    for (int kt = 0; kt < 7; ++kt) {
        v16h a = load_a_frag(nf, KPN_, kt * 32);
#pragma unroll
        for (int nt = 0; nt < 8; ++nt) {
            v16h bb = load_b_frag(wn1f + (size_t)(kt * 8 + nt) * 512);
            acc[nt] = wmma_f16(a, bb, acc[nt]);
        }
    }
#pragma unroll
    for (int nt = 0; nt < 8; ++nt) {
        float bv = bn1[nt * 16 + nlo];
#pragma unroll
        for (int r = 0; r < 8; ++r)
            nf[(r + 8 * hi) * KPN_ + nt * 16 + nlo] = (_Float16)siluf(acc[nt][r] + bv);
    }

    // ---- GEMM: h @ Wn2 (128 -> 80 padded)
    v8f acc2[5];
#pragma unroll
    for (int nt = 0; nt < 5; ++nt)
#pragma unroll
        for (int i = 0; i < 8; ++i) acc2[nt][i] = 0.0f;
#pragma unroll
    for (int kt = 0; kt < 4; ++kt) {
        v16h a = load_a_frag(nf, KPN_, kt * 32);
#pragma unroll
        for (int nt = 0; nt < 5; ++nt) {
            v16h bb = load_b_frag(wn2f + (size_t)(kt * 5 + nt) * 512);
            acc2[nt] = wmma_f16(a, bb, acc2[nt]);
        }
    }
    // residual update of hydrogen features
#pragma unroll
    for (int nt = 0; nt < 5; ++nt) {
#pragma unroll
        for (int r = 0; r < 8; ++r) {
            int n = nt * 16 + nlo;
            int node = base + r + 8 * hi;
            if (n < FDIM_ && node < H_) {
                float oldv = (float)feat_bs[(size_t)node * FSTR_ + n];
                feat_bs[(size_t)node * FSTR_ + n] = (_Float16)(oldv + acc2[nt][r] + bn2[n]);
            }
        }
    }
    // position update
    if (lane < 16) {
        int node = base + lane;
        if (node < H_) {
#pragma unroll
            for (int d = 0; d < 3; ++d)
                x_bs[(size_t)node * 3 + d] += aggx_bs[(size_t)node * 3 + d];
        }
    }
}

// ---------------------------------------------------------------------------
__global__ void finalize_kernel(const float* __restrict__ x, const float* __restrict__ x0,
                                const float* __restrict__ amask, float* __restrict__ out) {
    int idx = blockIdx.x * blockDim.x + threadIdx.x;
    if (idx >= B_ * S_ * H_ * 3) return;
    int d = idx % 3;
    int h = (idx / 3) % H_;
    int bs = idx / (3 * H_);
    int b = bs / S_;
    out[idx] = (x[((size_t)bs * NNODE_ + h) * 3 + d] - x0[idx]) * amask[b * H_ + h];
}

// ---------------------------------------------------------------------------
extern "C" void kernel_launch(void* const* d_in, const int* in_sizes, int n_in,
                              void* d_out, int out_size, void* d_ws, size_t ws_size,
                              hipStream_t stream) {
    (void)in_sizes; (void)n_in; (void)out_size; (void)ws_size;
    const float* t       = (const float*)d_in[0];
    const float* x_h     = (const float*)d_in[1];
    const float* x_heavy = (const float*)d_in[2];
    const float* bond_hm = (const float*)d_in[3];
    const float* em_hv   = (const float*)d_in[4];
    const float* em_hh   = (const float*)d_in[5];
    const float* amask   = (const float*)d_in[6];
    const float* W1  = (const float*)d_in[7];
    const float* b1  = (const float*)d_in[8];
    const float* W2  = (const float*)d_in[9];
    const float* b2  = (const float*)d_in[10];
    const float* Wc1 = (const float*)d_in[11];
    const float* bc1 = (const float*)d_in[12];
    const float* Wc2 = (const float*)d_in[13];
    const float* Wn1 = (const float*)d_in[14];
    const float* bn1 = (const float*)d_in[15];
    const float* Wn2 = (const float*)d_in[16];
    const float* bn2 = (const float*)d_in[17];
    const int* pep    = (const int*)d_in[18];
    const int* lab_hv = (const int*)d_in[19];
    const int* lab_h  = (const int*)d_in[20];
    const int* pos_hv = (const int*)d_in[21];
    const int* pos_h  = (const int*)d_in[22];
    const int* ehh    = (const int*)d_in[23];
    const int* bound  = (const int*)d_in[24];

    char* ws = (char*)d_ws;
    size_t off = 0;
    auto alloc = [&](size_t bytes) {
        size_t o = off;
        off += (bytes + 255) & ~(size_t)255;
        return o;
    };
    const size_t feat_b = (size_t)B_ * S_ * NNODE_ * FSTR_ * 2;
    const size_t x_b    = (size_t)B_ * S_ * NNODE_ * 3 * 4;
    const size_t x0_b   = (size_t)B_ * S_ * H_ * 3 * 4;
    const size_t ei_b   = (size_t)B_ * E_ * 4;
    const size_t aggm_b = (size_t)B_ * S_ * H_ * HID_ * 4;
    const size_t aggx_b = (size_t)B_ * S_ * H_ * 3 * 4;

    size_t o_feat = alloc(feat_b);
    size_t o_x    = alloc(x_b);
    size_t o_x0   = alloc(x0_b);
    size_t o_erow = alloc(ei_b);
    size_t o_ecol = alloc(ei_b);
    size_t o_ebnd = alloc(ei_b);
    size_t o_emsk = alloc(ei_b);
    size_t o_aggm = alloc(aggm_b + aggx_b);   // contiguous so one memset clears both
    size_t o_aggx = o_aggm + aggm_b;
    size_t o_w1f[L_], o_w2f[L_], o_wc1f[L_], o_wn1f[L_], o_wn2f[L_];
    for (int l = 0; l < L_; ++l) {
        o_w1f[l]  = alloc((size_t)48 * 512 * 2);   // 6 kt x 8 nt
        o_w2f[l]  = alloc((size_t)32 * 512 * 2);   // 4 x 8
        o_wc1f[l] = alloc((size_t)32 * 512 * 2);   // 4 x 8
        o_wn1f[l] = alloc((size_t)56 * 512 * 2);   // 7 x 8
        o_wn2f[l] = alloc((size_t)20 * 512 * 2);   // 4 x 5
    }

    _Float16* featp = (_Float16*)(ws + o_feat);
    float* xp  = (float*)(ws + o_x);
    float* x0p = (float*)(ws + o_x0);
    int* erowp = (int*)(ws + o_erow);
    int* ecolp = (int*)(ws + o_ecol);
    float* ebndp = (float*)(ws + o_ebnd);
    float* emskp = (float*)(ws + o_emsk);
    float* aggmp = (float*)(ws + o_aggm);
    float* aggxp = (float*)(ws + o_aggx);

    // ---- weight swizzles (per layer)
    for (int l = 0; l < L_; ++l) {
        swizzle_weights<<<48, 32, 0, stream>>>(W1 + (size_t)l * EDIM_ * HID_,
            (_Float16*)(ws + o_w1f[l]), 8, HID_, HID_, 1, EDIM_);
        swizzle_weights<<<32, 32, 0, stream>>>(W2 + (size_t)l * HID_ * HID_,
            (_Float16*)(ws + o_w2f[l]), 8, HID_, HID_, 0, HID_);
        swizzle_weights<<<32, 32, 0, stream>>>(Wc1 + (size_t)l * HID_ * HID_,
            (_Float16*)(ws + o_wc1f[l]), 8, HID_, HID_, 0, HID_);
        swizzle_weights<<<56, 32, 0, stream>>>(Wn1 + (size_t)l * (FDIM_ + HID_) * HID_,
            (_Float16*)(ws + o_wn1f[l]), 8, HID_, HID_, 2, FDIM_ + HID_);
        swizzle_weights<<<20, 32, 0, stream>>>(Wn2 + (size_t)l * HID_ * FDIM_,
            (_Float16*)(ws + o_wn2f[l]), 5, FDIM_, FDIM_, 0, HID_);
    }

    // ---- graph/feature prep
    prep_feat<<<(B_ * NNODE_ + 127) / 128, 128, 0, stream>>>(lab_h, lab_hv, pos_h, pos_hv,
                                                             pep, featp);
    prep_edges<<<(B_ * E_ + 255) / 256, 256, 0, stream>>>(ehh, em_hh, em_hv, bond_hm,
                                                          erowp, ecolp, ebndp, emskp);
    prep_x<<<(B_ * S_ * NNODE_ + 255) / 256, 256, 0, stream>>>(x_h, x_heavy, bound, xp, x0p);

    // ---- layer loop
    for (int l = 0; l < L_; ++l) {
        hipMemsetAsync(ws + o_aggm, 0, aggm_b + aggx_b, stream);
        edge_kernel<<<dim3((E_ / 16 + 7) / 8, B_ * S_), 256, 0, stream>>>(
            featp, xp, erowp, ecolp, ebndp, emskp, t,
            (const _Float16*)(ws + o_w1f[l]), (const _Float16*)(ws + o_w2f[l]),
            (const _Float16*)(ws + o_wc1f[l]),
            b1 + (size_t)l * HID_, b2 + (size_t)l * HID_,
            bc1 + (size_t)l * HID_, Wc2 + (size_t)l * HID_,
            aggmp, aggxp);
        node_kernel<<<B_ * S_, 128, 0, stream>>>(
            featp, xp, aggmp, aggxp,
            (const _Float16*)(ws + o_wn1f[l]), (const _Float16*)(ws + o_wn2f[l]),
            bn1 + (size_t)l * HID_, bn2 + (size_t)l * FDIM_);
    }

    finalize_kernel<<<(B_ * S_ * H_ * 3 + 255) / 256, 256, 0, stream>>>(
        xp, x0p, amask, (float*)d_out);
}
